// SeqPred_6425271075221
// MI455X (gfx1250) — matrix operand order repaired and checked by reading, not compiled, not run on hardware
//
#include <hip/hip_runtime.h>
#include <hip/hip_bf16.h>
#include <math.h>

// ---------------------------------------------------------------------------
// SeqPred pipeline for MI455X (gfx1250, wave32, WMMA).
//  dense1 -> BN(train stats) -> relu -> concat -> BiGRU -> dense(relu)
//        -> emissions f / transitions g -> CRF fwd/bwd scans.
//
// All heavy GEMMs: v_wmma_f32_16x16x32_f16, fragments loaded DIRECTLY from
// row-major f16 global memory as two b128 loads per lane (no LDS staging,
// no barriers) -- operands are pre-converted/zero-padded to K%32==0 once.
// GRU recurrence: persistent kernel, h double-buffered in LDS (f16), three
// gate WMMA accumulators fused with sigmoid/tanh epilogue in registers.
//
// Workspace requirement: ~547 MB (layout in kernel_launch).
// ---------------------------------------------------------------------------

typedef __attribute__((ext_vector_type(16))) _Float16 v16h;
typedef __attribute__((ext_vector_type(8)))  float    v8f;
typedef __attribute__((ext_vector_type(4)))  unsigned u32x4;

#define Bn    128
#define Tn    512
#define DIN   42
#define H1n   200
#define H2n   200
#define HGRU  400
#define G3    1200      // 3*HGRU
#define KPAD  416       // HGRU padded to multiple of 32
#define KCH   13        // KPAD/32
#define NCT   25        // HGRU/16 column tiles
#define MROWS (Bn*Tn)   // 65536
#define KCRF  8
#define EPSC  1e-5f

union Frag { v16h v; unsigned u[8]; u32x4 q[2]; _Float16 h[16]; };

// ---------------------------------------------------------------------------
// Direct-load WMMA GEMM:  C[M,N] = act( A[M,Kp] * W[N,Kp]^T + bias )
// A, W: f16 row-major, K zero-padded to Kp (multiple of 32).
// M must be a multiple of 128. N arbitrary (B rows clamped; W pads are zero
// so clamped/garbage lanes only feed discarded accumulator columns).
// Block 256 thr = 8 waves; tile BM=128 BN=64; each wave 32x32 = 2x2 WMMA.
// ---------------------------------------------------------------------------
template <typename CT, bool RELU>
__global__ __launch_bounds__(256)
void wmma_gemm_h(const _Float16* __restrict__ A, const _Float16* __restrict__ W,
                 const float* __restrict__ bias, CT* __restrict__ C,
                 int M, int N, int Kp, int ldc)
{
    const int tid  = threadIdx.x;
    const int lane = tid & 31;
    const int wave = tid >> 5;
    const int wr   = wave >> 1;           // 0..3 -> 32-row strip
    const int wc   = wave & 1;            // 0..1 -> 32-col strip
    const int grp  = lane >> 4;           // K-half select
    const int l15  = lane & 15;

    const int row0 = blockIdx.y * 128 + wr * 32 + l15;        // A row, frag su=0
    const int row1 = row0 + 16;                               // A row, frag su=1
    const int c0   = blockIdx.x * 64 + wc * 32 + l15;         // B col, frag sv=0
    const int c1   = c0 + 16;                                 // B col, frag sv=1
    const int n0   = min(c0, N - 1);                          // clamp: pads are 0
    const int n1   = min(c1, N - 1);

    // per-lane fragment base pointers (16B-aligned: Kp%32==0)
    const u32x4* ap0 = (const u32x4*)(A + (size_t)row0 * Kp + grp * 16);
    const u32x4* ap1 = (const u32x4*)(A + (size_t)row1 * Kp + grp * 16);
    const u32x4* bp0 = (const u32x4*)(W + (size_t)n0   * Kp + grp * 16);
    const u32x4* bp1 = (const u32x4*)(W + (size_t)n1   * Kp + grp * 16);

    const v8f vzero = {0.f,0.f,0.f,0.f,0.f,0.f,0.f,0.f};
    v8f acc00 = vzero, acc01 = vzero, acc10 = vzero, acc11 = vzero;

    const int nk = Kp >> 5;
    for (int kc = 0; kc < nk; ++kc) {
        // A frag: lane = row l15, 16 contiguous halves at grp*16; B frag: 16
        // contiguous halves at grp*16. Each = two global_load_b128.
        Frag a0, a1, b0, b1;
        const int qa = kc * 4;            // 32 halfs = 4 u32x4 per chunk
        a0.q[0] = ap0[qa];     a0.q[1] = ap0[qa + 1];
        a1.q[0] = ap1[qa];     a1.q[1] = ap1[qa + 1];
        b0.q[0] = bp0[qa];     b0.q[1] = bp0[qa + 1];
        b1.q[0] = bp1[qa];     b1.q[1] = bp1[qa + 1];
        if (kc + 1 < nk) {                       // global_prefetch_b8
            __builtin_prefetch((const char*)(ap0 + qa + 4), 0, 1);
            __builtin_prefetch((const char*)(bp0 + qa + 4), 0, 1);
        }
        acc00 = __builtin_amdgcn_wmma_f32_16x16x32_f16(false, a0.v, false, b0.v, (short)0, acc00, false, false);
        acc01 = __builtin_amdgcn_wmma_f32_16x16x32_f16(false, a0.v, false, b1.v, (short)0, acc01, false, false);
        acc10 = __builtin_amdgcn_wmma_f32_16x16x32_f16(false, a1.v, false, b0.v, (short)0, acc10, false, false);
        acc11 = __builtin_amdgcn_wmma_f32_16x16x32_f16(false, a1.v, false, b1.v, (short)0, acc11, false, false);
    }

    // epilogue: C layout VGPR v -> M = v + 8*grp, N = lane%16
    const int erow0 = blockIdx.y * 128 + wr * 32 + grp * 8;
    #pragma unroll
    for (int v = 0; v < 8; ++v) {
        int r0 = erow0 + v, r1 = r0 + 16;
        float x00 = acc00[v], x01 = acc01[v], x10 = acc10[v], x11 = acc11[v];
        if (bias) {
            if (c0 < N) { x00 += bias[c0]; x10 += bias[c0]; }
            if (c1 < N) { x01 += bias[c1]; x11 += bias[c1]; }
        }
        if (RELU) {
            x00 = fmaxf(x00, 0.f); x01 = fmaxf(x01, 0.f);
            x10 = fmaxf(x10, 0.f); x11 = fmaxf(x11, 0.f);
        }
        if (c0 < N) { C[(size_t)r0 * ldc + c0] = (CT)x00; C[(size_t)r1 * ldc + c0] = (CT)x10; }
        if (c1 < N) { C[(size_t)r0 * ldc + c1] = (CT)x01; C[(size_t)r1 * ldc + c1] = (CT)x11; }
    }
}

// ---------------------------------------------------------------------------
// Persistent BiGRU recurrence. grid = 16: blockIdx/8 = dir, blockIdx%8 = batch
// group of 16 rows. h double-buffered in LDS (f16, K padded to 416). Each
// wave owns 16x16 gate-column tiles; r/z/n WMMA accumulation fused with the
// gate nonlinearity + h update in registers. One barrier per time step.
// ---------------------------------------------------------------------------
__global__ __launch_bounds__(256)
void gru_persist(const _Float16* __restrict__ giF, const _Float16* __restrict__ giB,
                 const _Float16* __restrict__ WpF, const _Float16* __restrict__ WpB,
                 const float* __restrict__ bhhF, const float* __restrict__ bhhB,
                 _Float16* __restrict__ hcat)
{
    __shared__ _Float16 hsh[2][16][KPAD];

    const int tid  = threadIdx.x;
    const int lane = tid & 31;
    const int wave = tid >> 5;
    const int dir  = blockIdx.x >> 3;
    const int bg   = blockIdx.x & 7;
    const _Float16* gi  = dir ? giB : giF;
    const _Float16* Wp  = dir ? WpB : WpF;
    const float*    bhh = dir ? bhhB : bhhF;
    const int grp = lane >> 4;
    const int l15 = lane & 15;
    const v8f vzero = {0.f,0.f,0.f,0.f,0.f,0.f,0.f,0.f};

    for (int i = tid; i < 2 * 16 * KPAD; i += 256) (&hsh[0][0][0])[i] = (_Float16)0.f;
    __syncthreads();

    for (int s = 0; s < Tn; ++s) {
        const int t   = dir ? (Tn - 1 - s) : s;
        const int cur = s & 1, nxt = cur ^ 1;
        const u32x4* hq = (const u32x4*)(&hsh[cur][l15][0]);   // 832B rows, 16B ok

        for (int ct = wave; ct < NCT; ct += 8) {       // wave-uniform loop
            const int ncol = ct * 16 + l15;            // gate column (0..399)
            const u32x4* wrq = (const u32x4*)(Wp + (size_t)(      ncol) * KPAD + grp * 16);
            const u32x4* wzq = (const u32x4*)(Wp + (size_t)(400 + ncol) * KPAD + grp * 16);
            const u32x4* wnq = (const u32x4*)(Wp + (size_t)(800 + ncol) * KPAD + grp * 16);
            v8f ar = vzero, az = vzero, an = vzero;

            for (int kc = 0; kc < KCH; ++kc) {
                Frag a, br, bz, bn2;
                const int qa = kc * 4;
                a.q[0]   = hq[qa + 2 * grp];  a.q[1]   = hq[qa + 2 * grp + 1];  // ds_load_b128 x2
                br.q[0]  = wrq[qa];       br.q[1]  = wrq[qa + 1];
                bz.q[0]  = wzq[qa];       bz.q[1]  = wzq[qa + 1];
                bn2.q[0] = wnq[qa];       bn2.q[1] = wnq[qa + 1];
                ar = __builtin_amdgcn_wmma_f32_16x16x32_f16(false, a.v, false, br.v,  (short)0, ar, false, false);
                az = __builtin_amdgcn_wmma_f32_16x16x32_f16(false, a.v, false, bz.v,  (short)0, az, false, false);
                an = __builtin_amdgcn_wmma_f32_16x16x32_f16(false, a.v, false, bn2.v, (short)0, an, false, false);
            }

            // fused gate epilogue (PyTorch order r,z,n)
            #pragma unroll
            for (int v = 0; v < 8; ++v) {
                int lrow = v + grp * 8;                  // 0..15
                int b    = bg * 16 + lrow;
                int j    = ncol;
                size_t gb = ((size_t)t * Bn + b) * (size_t)G3;
                float grv = (float)gi[gb + j]       + ar[v] + bhh[j];
                float gzv = (float)gi[gb + 400 + j] + az[v] + bhh[400 + j];
                float hnv =                           an[v] + bhh[800 + j];
                float r = 1.f / (1.f + expf(-grv));
                float z = 1.f / (1.f + expf(-gzv));
                float n = tanhf((float)gi[gb + 800 + j] + r * hnv);
                float hp = (float)hsh[cur][lrow][j];
                float hn = (1.f - z) * n + z * hp;
                hsh[nxt][lrow][j] = (_Float16)hn;
                hcat[((size_t)t * Bn + b) * 800 + (size_t)dir * 400 + j] = (_Float16)hn;
            }
        }
        __syncthreads();   // publish hsh[nxt] before next step reads it
    }
}

// ---------------------------------------------------------------------------
// Prep / BN / misc kernels
// ---------------------------------------------------------------------------
// f32 (rows,K) -> f16 (rows,Kp) zero-padded
__global__ void pad_cvt(const float* __restrict__ src, _Float16* __restrict__ dst,
                        int rows, int K, int Kp)
{
    size_t idx = (size_t)blockIdx.x * blockDim.x + threadIdx.x;
    if (idx >= (size_t)rows * Kp) return;
    int k = (int)(idx % Kp);
    size_t n = idx / Kp;
    dst[idx] = (k < K) ? (_Float16)src[n * K + k] : (_Float16)0.f;
}

__global__ void zero_f32(float* p, int n)
{
    int i = blockIdx.x * blockDim.x + threadIdx.x;
    if (i < n) p[i] = 0.f;
}

__global__ void bn_stats(const float* __restrict__ x1, float* __restrict__ stats)
{
    const int c = threadIdx.x;
    if (c >= H1n) return;
    const size_t r0 = (size_t)blockIdx.x * 1024;
    float s = 0.f, q = 0.f;
    for (int r = 0; r < 1024; ++r) {
        float v = x1[(r0 + r) * H1n + c];
        s += v; q += v * v;
    }
    atomicAdd(&stats[c], s);
    atomicAdd(&stats[H1n + c], q);
}

// (B,T) order in -> (T,B) order out, concat [inp | bn_relu(x1)], f16,
// row stride 256 with zeroed pad columns (GEMM-A K-padding).
__global__ void bn_apply_concat(const float* __restrict__ inp, const float* __restrict__ x1,
                                const float* __restrict__ stats, const float* __restrict__ gamma,
                                const float* __restrict__ beta, _Float16* __restrict__ xcat)
{
    size_t idx = (size_t)blockIdx.x * blockDim.x + threadIdx.x;
    if (idx >= (size_t)MROWS * 256) return;
    int c = (int)(idx % 256);
    size_t m = idx / 256;                 // m = b*T + t
    int b = (int)(m / Tn), t = (int)(m % Tn);
    float val = 0.f;
    if (c < DIN) {
        val = inp[m * DIN + c];
    } else if (c < DIN + H1n) {
        int ch = c - DIN;
        float mu  = stats[ch] * (1.f / MROWS);
        float var = stats[H1n + ch] * (1.f / MROWS) - mu * mu;
        float xv  = x1[m * H1n + ch];
        val = (xv - mu) * rsqrtf(var + EPSC) * gamma[ch] + beta[ch];
        val = fmaxf(val, 0.f);
    }
    xcat[((size_t)t * Bn + b) * 256 + c] = (_Float16)val;
}

// ---------------------------------------------------------------------------
// CRF forward/backward scans (K=8). One thread per batch element.
// blockIdx.x==0: forward (nu_alp). blockIdx.x==1: backward (nu_bet).
// ---------------------------------------------------------------------------
__global__ __launch_bounds__(128)
void crf_scan(const float* __restrict__ f, const float* __restrict__ g,
              const float* __restrict__ mask, float* __restrict__ nua,
              float* __restrict__ nub)
{
    const int b = threadIdx.x;
    float nu[KCRF], nn[KCRF], gt[64];

    if (blockIdx.x == 0) {
        #pragma unroll
        for (int j = 0; j < KCRF; ++j) { nu[j] = f[(size_t)b * 8 + j]; nua[(size_t)b * 8 + j] = nu[j]; }
        for (int t = 1; t < Tn; ++t) {
            size_t r = (size_t)t * Bn + b;
            for (int i = 0; i < 64; ++i) gt[i] = g[r * 64 + i];
            float m = mask[r];
            #pragma unroll
            for (int j = 0; j < KCRF; ++j) {
                float mx = -1e30f;
                for (int i = 0; i < KCRF; ++i) mx = fmaxf(mx, gt[i * 8 + j] + nu[i]);
                float sm = 0.f;
                for (int i = 0; i < KCRF; ++i) sm += expf(gt[i * 8 + j] + nu[i] - mx);
                nn[j] = f[r * 8 + j] + mx + logf(sm);
            }
            #pragma unroll
            for (int j = 0; j < KCRF; ++j) { nu[j] = m * nn[j] + (1.f - m) * nu[j]; nua[r * 8 + j] = nu[j]; }
        }
    } else {
        #pragma unroll
        for (int j = 0; j < KCRF; ++j) { nu[j] = 0.f; nub[((size_t)(Tn - 1) * Bn + b) * 8 + j] = 0.f; }
        for (int t = Tn - 2; t >= 0; --t) {
            size_t r1 = (size_t)(t + 1) * Bn + b;
            for (int i = 0; i < 64; ++i) gt[i] = g[r1 * 64 + i];
            float tmp[KCRF];
            #pragma unroll
            for (int j = 0; j < KCRF; ++j) tmp[j] = f[r1 * 8 + j] + nu[j];
            float m = mask[r1];
            #pragma unroll
            for (int i = 0; i < KCRF; ++i) {
                float mx = -1e30f;
                for (int j = 0; j < KCRF; ++j) mx = fmaxf(mx, gt[i * 8 + j] + tmp[j]);
                float sm = 0.f;
                for (int j = 0; j < KCRF; ++j) sm += expf(gt[i * 8 + j] + tmp[j] - mx);
                nn[i] = mx + logf(sm);
            }
            #pragma unroll
            for (int i = 0; i < KCRF; ++i) { nu[i] = m * nn[i] + (1.f - m) * nu[i]; nub[((size_t)t * Bn + b) * 8 + i] = nu[i]; }
        }
    }
}

// ---------------------------------------------------------------------------
extern "C" void kernel_launch(void* const* d_in, const int* in_sizes, int n_in,
                              void* d_out, int out_size, void* d_ws, size_t ws_size,
                              hipStream_t stream)
{
    (void)in_sizes; (void)n_in; (void)out_size; (void)ws_size;

    const float* inp   = (const float*)d_in[0];
    /* d_in[1] seq_lengths: encoded in mask */
    const float* mask  = (const float*)d_in[2];
    const float* w1    = (const float*)d_in[3];
    const float* gamma = (const float*)d_in[4];
    const float* beta  = (const float*)d_in[5];
    const float* Wih_f = (const float*)d_in[6];
    const float* Whh_f = (const float*)d_in[7];
    const float* bih_f = (const float*)d_in[8];
    const float* bhh_f = (const float*)d_in[9];
    const float* Wih_b = (const float*)d_in[10];
    const float* Whh_b = (const float*)d_in[11];
    const float* bih_b = (const float*)d_in[12];
    const float* bhh_b = (const float*)d_in[13];
    const float* w2    = (const float*)d_in[14];
    const float* b2    = (const float*)d_in[15];
    const float* w3    = (const float*)d_in[16];
    const float* b3    = (const float*)d_in[17];
    const float* wl    = (const float*)d_in[18];
    const float* bl    = (const float*)d_in[19];

    // workspace layout (~547 MB total, offsets 256B-aligned)
    char* p = (char*)d_ws;
    float*    x1    = (float*)p;    p += (size_t)MROWS * H1n * 4;    // 52.4 MB
    float*    stats = (float*)p;    p += 2048;
    _Float16* inp_h = (_Float16*)p; p += (size_t)MROWS * 64 * 2;     // 8.4 MB  (42->64)
    _Float16* xcat  = (_Float16*)p; p += (size_t)MROWS * 256 * 2;    // 33.6 MB (242->256)
    _Float16* w1h   = (_Float16*)p; p += (size_t)H1n * 64 * 2;
    _Float16* WihFh = (_Float16*)p; p += (size_t)G3 * 256 * 2;
    _Float16* WihBh = (_Float16*)p; p += (size_t)G3 * 256 * 2;
    _Float16* WpF   = (_Float16*)p; p += (size_t)G3 * KPAD * 2;
    _Float16* WpB   = (_Float16*)p; p += (size_t)G3 * KPAD * 2;
    _Float16* w2h   = (_Float16*)p; p += (size_t)H2n * 800 * 2;
    _Float16* w3h   = (_Float16*)p; p += (size_t)64 * 224 * 2;
    _Float16* wlh   = (_Float16*)p; p += (size_t)KCRF * 224 * 2;
    _Float16* giFp  = (_Float16*)p; p += (size_t)MROWS * G3 * 2;     // 157.3 MB
    _Float16* giBp  = (_Float16*)p; p += (size_t)MROWS * G3 * 2;     // 157.3 MB
    _Float16* hcat  = (_Float16*)p; p += (size_t)MROWS * 800 * 2;    // 104.9 MB
    _Float16* out2  = (_Float16*)p; p += (size_t)MROWS * 224 * 2;    // 29.4 MB (200->224)

    // output regions: concat(f, g, nu_alp, nu_bet), all (T,B,...) t-major
    float* f_out = (float*)d_out;
    float* g_out = f_out + (size_t)MROWS * KCRF;
    float* nua   = g_out + (size_t)MROWS * 64;
    float* nub   = nua   + (size_t)MROWS * KCRF;

    dim3 blk(256);
    auto gcvt = [](size_t n) { return (unsigned)((n + 255) / 256); };

    // 0) one-time f32 -> f16 zero-padded operand prep
    pad_cvt<<<gcvt((size_t)MROWS * 64), blk, 0, stream>>>(inp,   inp_h, MROWS, DIN, 64);
    pad_cvt<<<gcvt((size_t)H1n * 64),   blk, 0, stream>>>(w1,    w1h,   H1n,  DIN, 64);
    pad_cvt<<<gcvt((size_t)G3 * 256),   blk, 0, stream>>>(Wih_f, WihFh, G3,   242, 256);
    pad_cvt<<<gcvt((size_t)G3 * 256),   blk, 0, stream>>>(Wih_b, WihBh, G3,   242, 256);
    pad_cvt<<<gcvt((size_t)G3 * KPAD),  blk, 0, stream>>>(Whh_f, WpF,   G3,   HGRU, KPAD);
    pad_cvt<<<gcvt((size_t)G3 * KPAD),  blk, 0, stream>>>(Whh_b, WpB,   G3,   HGRU, KPAD);
    pad_cvt<<<gcvt((size_t)H2n * 800),  blk, 0, stream>>>(w2,    w2h,   H2n,  800, 800);
    pad_cvt<<<gcvt((size_t)64 * 224),   blk, 0, stream>>>(w3,    w3h,   64,   H2n, 224);
    pad_cvt<<<gcvt((size_t)KCRF * 224), blk, 0, stream>>>(wl,    wlh,   KCRF, H2n, 224);

    // 1) dense1: x1 = inp @ w1^T  (65536 x 200, Kp=64) -> f32 for BN stats
    wmma_gemm_h<float, false><<<dim3((H1n + 63) / 64, MROWS / 128), blk, 0, stream>>>(
        inp_h, w1h, nullptr, x1, MROWS, H1n, 64, H1n);

    // 2) BN training-mode stats over all B*T rows
    zero_f32<<<2, 256, 0, stream>>>(stats, 2 * H1n);
    bn_stats<<<64, 256, 0, stream>>>(x1, stats);

    // 3) BN apply + relu + concat [inp|x1] -> (T,B,256-padded) f16
    bn_apply_concat<<<gcvt((size_t)MROWS * 256), blk, 0, stream>>>(
        inp, x1, stats, gamma, beta, xcat);

    // 4) hoisted GRU input projections: gi = xcat @ Wih^T + bih (65536x1200)
    wmma_gemm_h<_Float16, false><<<dim3((G3 + 63) / 64, MROWS / 128), blk, 0, stream>>>(
        xcat, WihFh, bih_f, giFp, MROWS, G3, 256, G3);
    wmma_gemm_h<_Float16, false><<<dim3((G3 + 63) / 64, MROWS / 128), blk, 0, stream>>>(
        xcat, WihBh, bih_b, giBp, MROWS, G3, 256, G3);

    // 5) sequential BiGRU recurrence (persistent, 2 dirs x 8 batch groups)
    gru_persist<<<16, blk, 0, stream>>>(giFp, giBp, WpF, WpB, bhh_f, bhh_b, hcat);

    // 6) out = relu(hcat @ w2^T + b2)  (65536x200, Kp=800), ldc=224 for reuse
    wmma_gemm_h<_Float16, true><<<dim3((H2n + 63) / 64, MROWS / 128), blk, 0, stream>>>(
        hcat, w2h, b2, out2, MROWS, H2n, 800, 224);

    // 7) g = out @ w3^T + b3  (65536x64, Kp=224) -> d_out, f32
    wmma_gemm_h<float, false><<<dim3(1, MROWS / 128), blk, 0, stream>>>(
        out2, w3h, b3, g_out, MROWS, 64, 224, 64);

    // 8) f = out @ wl^T + bl  (65536x8, Kp=224) -> d_out, f32
    wmma_gemm_h<float, false><<<dim3(1, MROWS / 128), blk, 0, stream>>>(
        out2, wlh, bl, f_out, MROWS, KCRF, 224, KCRF);

    // 9) CRF forward + backward scans
    crf_scan<<<2, 128, 0, stream>>>(f_out, g_out, mask, nua, nub);
}
